// EnhancedMultiHeadAttention_46351287058680
// MI455X (gfx1250) — matrix operand order repaired and checked by reading, not compiled, not run on hardware
//
#include <hip/hip_runtime.h>
#include <hip/hip_bf16.h>

// ---------------------------------------------------------------------------
// EnhancedMultiHeadAttention for MI455X (gfx1250, wave32, WMMA).
//
// Shapes: B=4, L=2048, E=512, H=8.  M = B*L = 8192, proj width N = H*E = 4096.
// Roofline: ~138 GFLOP vs ~0.5-1 GB HBM traffic @23.3TB/s => bf16 WMMA
// (16x16x32) makes compute (~30us) match memory time; fp32 WMMA (16x16x4)
// would be ~8x slower per instruction and become the bottleneck.
// Weights/activations are L2-resident (192MB L2), so K-tile re-reads are cheap.
//
// Pipeline (all on `stream`, workspace = 256 MB of d_ws):
//   [1] wmma_gemm<f32A, bf16OUT> x3 : Q/K/V = bf16(X @ W + b)   (8192x4096)
//   [2] attn_kernel : per-position 8x8 head-gram, masked softmax over 8,
//       attn @ V, written bf16 in the reference's raw-reshape-permuted layout
//   [3] wmma_gemm<bf16A, f32OUT+bias> : d_out = A2 @ Wo + bo    (8192x512)
//
// Epilogue stages the C tile through LDS so global writes are coalesced
// 16B-per-thread streams instead of 64 strided scalar stores per lane.
// ---------------------------------------------------------------------------

typedef __bf16 bf16_t;
typedef __attribute__((ext_vector_type(16))) __bf16 bf16x16;
typedef __attribute__((ext_vector_type(8)))  __bf16 bf16x8;
typedef __attribute__((ext_vector_type(4)))  __bf16 bf16x4;
typedef __attribute__((ext_vector_type(8)))  float  f32x8;
typedef __attribute__((ext_vector_type(4)))  float  f32x4;

#define BM 128
#define BN 128
#define BK 32
#define LDT 40        // padded LDS row stride (halves); 80B keeps 16B alignment
#define SMEM_BYTES 40960

// ---------------------------------------------------------------------------
// Tiled bf16 WMMA GEMM:  C[M x N] = A[M x K] * B[K x N] + bias[N]
//   A: fp32 (converted in-flight) or bf16 (template).
//   B: fp32 weights, converted to bf16 into LDS, stored transposed ([n][k]) so
//      a B fragment is two contiguous 16-byte ds reads matching the V_WMMA
//      16-bit B layout (lane = column n, K = {hi*8.., 16+hi*8..} per lane half-group).
//   A LDS is row-major; an A fragment is two 16-byte ds reads at k-offsets
//      {hi*8, 16+hi*8} matching the 16-bit A VGPR layout.
// Grid: (N/128, M/128), 256 threads = 8 waves (4 along M x 2 along N);
// each wave owns a 32x64 patch = 2x4 WMMA accumulators, K stepped by 32
// with double-buffered LDS and global_prefetch_b8 two K-tiles ahead.
// ---------------------------------------------------------------------------
template <bool A_F32, bool OUT_F32>
__global__ __launch_bounds__(256)
void wmma_gemm(const void* __restrict__ Ap, const float* __restrict__ Bp,
               const float* __restrict__ bias, void* __restrict__ Cp,
               int K, int N)
{
    __shared__ __align__(16) unsigned char smem[SMEM_BYTES];
    auto* As = (bf16_t(*)[BM][LDT])smem;                        // [2][128][40]
    auto* Bs = (bf16_t(*)[BN][LDT])(smem + 2 * BM * LDT * 2);   // [2][128][40]

    const int t    = threadIdx.x;
    const int lane = t & 31;
    const int wave = t >> 5;
    const int wm   = wave >> 1;          // 0..3 -> m offset wm*32
    const int wn   = wave & 1;           // 0..1 -> n offset wn*64
    const int lr   = lane & 15;
    const int hi   = lane >> 4;          // lane half-group (ISA fragment layout)

    const long m0 = (long)blockIdx.y * BM;
    const long n0 = (long)blockIdx.x * BN;

    const float*  Af = (const float*)Ap;
    const bf16_t* Ab = (const bf16_t*)Ap;

    f32x8 acc[2][4];
    const f32x8 zero = {0.f,0.f,0.f,0.f,0.f,0.f,0.f,0.f};
#pragma unroll
    for (int mi = 0; mi < 2; mi++)
#pragma unroll
        for (int ni = 0; ni < 4; ni++) acc[mi][ni] = zero;

    auto load_tile = [&](int kt, int buf) {
        const long k0 = (long)kt * BK;
        // ---- A tile: 128 rows x 32 k ----
        {
            const int row  = t >> 1;
            const int colb = (t & 1) * 16;
            if constexpr (A_F32) {
                const float* src = Af + (m0 + row) * (long)K + k0 + colb;
#pragma unroll
                for (int q = 0; q < 4; q++) {
                    f32x4 v = *(const f32x4*)(src + q * 4);
#pragma unroll
                    for (int e = 0; e < 4; e++)
                        As[buf][row][colb + q * 4 + e] = (bf16_t)v[e];
                }
            } else {
                const bf16_t* src = Ab + (m0 + row) * (long)K + k0 + colb;
                bf16x8 v0 = *(const bf16x8*)src;
                bf16x8 v1 = *(const bf16x8*)(src + 8);
                *(bf16x8*)&As[buf][row][colb]     = v0;
                *(bf16x8*)&As[buf][row][colb + 8] = v1;
            }
        }
        // ---- B tile: 32 k x 128 n, stored transposed into LDS ----
        {
            const int nb = (t & 31) * 4;
#pragma unroll
            for (int p = 0; p < 4; p++) {
                const int kl = p * 8 + (t >> 5);
                const float* src = Bp + (k0 + kl) * (long)N + n0 + nb;
                f32x4 v = *(const f32x4*)src;
#pragma unroll
                for (int e = 0; e < 4; e++)
                    Bs[buf][nb + e][kl] = (bf16_t)v[e];
            }
        }
    };

    const int KT = K / BK;
    load_tile(0, 0);

    for (int kt = 0; kt < KT; kt++) {
        __syncthreads();
        if (kt + 1 < KT) {
            if (kt + 2 < KT) {   // global_prefetch_b8 two K-tiles ahead
                const long kpf = (long)(kt + 2) * BK;
                if constexpr (A_F32)
                    __builtin_prefetch(Af + (m0 + (t >> 1)) * (long)K + kpf, 0, 3);
                else
                    __builtin_prefetch(Ab + (m0 + (t >> 1)) * (long)K + kpf, 0, 3);
                __builtin_prefetch(Bp + kpf * (long)N + n0 + (t & 31) * 4, 0, 3);
            }
            load_tile(kt + 1, (kt + 1) & 1);
        }
        const int buf = kt & 1;

        bf16x16 afrag[2];
        bf16x16 bfrag[4];
#pragma unroll
        for (int mi = 0; mi < 2; mi++) {
            const bf16_t* ap = &As[buf][wm * 32 + mi * 16 + lr][hi * 8];
            bf16x8 lo = *(const bf16x8*)ap;          // K = hi*8 .. hi*8+7
            bf16x8 h8 = *(const bf16x8*)(ap + 16);   // K = 16+hi*8 .. +7
            afrag[mi] = __builtin_shufflevector(lo, h8,
                0,1,2,3,4,5,6,7,8,9,10,11,12,13,14,15);
        }
#pragma unroll
        for (int ni = 0; ni < 4; ni++) {
            const bf16_t* bp = &Bs[buf][wn * 64 + ni * 16 + lr][hi * 8];
            bf16x8 lo = *(const bf16x8*)bp;          // K = hi*8 .. hi*8+7
            bf16x8 h8 = *(const bf16x8*)(bp + 16);   // K = 16+hi*8 .. +7
            bfrag[ni] = __builtin_shufflevector(lo, h8,
                0,1,2,3,4,5,6,7,8,9,10,11,12,13,14,15);
        }
#pragma unroll
        for (int mi = 0; mi < 2; mi++)
#pragma unroll
            for (int ni = 0; ni < 4; ni++) {
#if defined(__HIP_DEVICE_COMPILE__)
                acc[mi][ni] = __builtin_amdgcn_wmma_f32_16x16x32_bf16(
                    false, afrag[mi], false, bfrag[ni],
                    (short)0, acc[mi][ni], false, false);
#endif
            }
    }

    // ---- Epilogue: bias add, stage C tile in LDS, coalesced global stores.
    // C fragment layout: lane owns column lr, rows r + 8*hi (r = VGPR index).
    __syncthreads();   // all waves done reading As/Bs; safe to reuse smem
    if constexpr (!OUT_F32) {
        bf16_t* Cs = (bf16_t*)smem;          // [128][136] (padded vs 64 banks)
        const int CLD = 136;
#pragma unroll
        for (int ni = 0; ni < 4; ni++) {
            const int cn  = wn * 64 + ni * 16 + lr;
            const float bvv = bias[n0 + cn];
#pragma unroll
            for (int mi = 0; mi < 2; mi++)
#pragma unroll
                for (int r = 0; r < 8; r++) {
                    const int cm = wm * 32 + mi * 16 + r + 8 * hi;
                    Cs[cm * CLD + cn] = (bf16_t)(acc[mi][ni][r] + bvv);
                }
        }
        __syncthreads();
        bf16_t* Cg = (bf16_t*)Cp;
        const int row  = t >> 1;
        const int colb = (t & 1) * 64;
#pragma unroll
        for (int q = 0; q < 8; q++) {
            bf16x8 v = *(const bf16x8*)&Cs[row * CLD + colb + q * 8];
            *(bf16x8*)&Cg[(m0 + row) * (long)N + n0 + colb + q * 8] = v;
        }
    } else {
        float* Cs = (float*)smem;            // [64][132], two 64-row passes
        const int CLD = 132;
        float* Cg = (float*)Cp;
#pragma unroll
        for (int h = 0; h < 2; h++) {
            if ((wm >> 1) == h) {
#pragma unroll
                for (int ni = 0; ni < 4; ni++) {
                    const int cn  = wn * 64 + ni * 16 + lr;
                    const float bvv = bias[n0 + cn];
#pragma unroll
                    for (int mi = 0; mi < 2; mi++)
#pragma unroll
                        for (int r = 0; r < 8; r++) {
                            const int cm = (wm & 1) * 32 + mi * 16 + r + 8 * hi;
                            Cs[cm * CLD + cn] = acc[mi][ni][r] + bvv;
                        }
                }
            }
            __syncthreads();
            const int row  = t >> 2;
            const int colb = (t & 3) * 32;
#pragma unroll
            for (int q = 0; q < 8; q++) {
                f32x4 v = *(const f32x4*)&Cs[row * CLD + colb + q * 4];
                *(f32x4*)&Cg[(m0 + h * 64 + row) * (long)N + n0 + colb + q * 4] = v;
            }
            __syncthreads();
        }
    }
}

// ---------------------------------------------------------------------------
// Per-position attention. One wave per (b,l); 8 positions per 256-thread WG.
//   energy[i][j] = sum_d Q[pos, i*512+d] * K[pos, j*512+d]   (64 pairs, 2/lane)
//   attn = softmax_j(mask ? energy/sqrt(512) : -inf)
//   O[i][d] = sum_j attn[i][j] * V[pos, j*512+d]
// Output written bf16 into the raw-reshape-permuted matrix A2[8192 x 4096]:
//   A2[b*2048 + i*256 + l/8,  (l%8)*512 + d] = O[i][d]
// so the output projection is a plain row-major GEMM.
// ---------------------------------------------------------------------------
__global__ __launch_bounds__(256)
void attn_kernel(const bf16_t* __restrict__ Q, const bf16_t* __restrict__ Km,
                 const bf16_t* __restrict__ V, const int* __restrict__ mask,
                 bf16_t* __restrict__ A2)
{
    __shared__ float sE[8][64];
    __shared__ float sA[8][64];

    const int  lane = threadIdx.x & 31;
    const int  wave = threadIdx.x >> 5;
    const long pos  = (long)blockIdx.x * 8 + wave;
    const long base = pos * 4096;

    // ---- energy: pairs p=lane (i=lane>>3, j=lane&7) and p+32 (i+4, same j)
    const int j  = lane & 7;
    const int i0 = lane >> 3;                 // 0..3
    float acc0 = 0.f, acc1 = 0.f;
    for (int d = 0; d < 512; d += 8) {
        bf16x8 kv = *(const bf16x8*)(Km + base + j * 512 + d);
        bf16x8 q0 = *(const bf16x8*)(Q  + base + i0 * 512 + d);
        bf16x8 q1 = *(const bf16x8*)(Q  + base + (i0 + 4) * 512 + d);
#pragma unroll
        for (int e = 0; e < 8; e++) {
            const float kf = (float)kv[e];
            acc0 += (float)q0[e] * kf;
            acc1 += (float)q1[e] * kf;
        }
    }
    sE[wave][lane]      = acc0;
    sE[wave][lane + 32] = acc1;
    __syncthreads();

    // ---- masked softmax over j (8 entries), lanes 0..7 own row i=lane
    if (lane < 8) {
        const float scale = 0.0441941738241592f;   // 1/sqrt(512)
        const int   i     = lane;
        float e[8];
        float mx = -1e30f;
#pragma unroll
        for (int jj = 0; jj < 8; jj++) {
            const int mk = mask[pos * 64 + i * 8 + jj];
            const float v = (mk == 0) ? -1e30f : sE[wave][i * 8 + jj] * scale;
            e[jj] = v;
            mx = fmaxf(mx, v);
        }
        float s = 0.f;
#pragma unroll
        for (int jj = 0; jj < 8; jj++) {
            e[jj] = __builtin_expf(e[jj] - mx);    // masked -> exp(-huge)=0
            s += e[jj];
        }
        const float inv = 1.f / s;
#pragma unroll
        for (int jj = 0; jj < 8; jj++) sA[wave][i * 8 + jj] = e[jj] * inv;
    }
    __syncthreads();

    // ---- O = attn @ V, permuted bf16 store
    float att[64];
#pragma unroll
    for (int tt = 0; tt < 64; tt++) att[tt] = sA[wave][tt];

    const int  b = (int)(pos >> 11);
    const int  l = (int)(pos & 2047);
    const long colBase = (long)(l & 7) * 512;
    const long rowBase = (long)b * 2048 + (l >> 3);

#pragma unroll
    for (int c = 0; c < 4; c++) {
        const int d0 = c * 128 + lane * 4;
        float vj[8][4];
#pragma unroll
        for (int jj = 0; jj < 8; jj++) {
            bf16x4 vv = *(const bf16x4*)(V + base + jj * 512 + d0);
#pragma unroll
            for (int e = 0; e < 4; e++) vj[jj][e] = (float)vv[e];
        }
#pragma unroll
        for (int i = 0; i < 8; i++) {
            float o0 = 0.f, o1 = 0.f, o2 = 0.f, o3 = 0.f;
#pragma unroll
            for (int jj = 0; jj < 8; jj++) {
                const float a = att[i * 8 + jj];
                o0 += a * vj[jj][0];
                o1 += a * vj[jj][1];
                o2 += a * vj[jj][2];
                o3 += a * vj[jj][3];
            }
            bf16x4 ov;
            ov[0] = (bf16_t)o0; ov[1] = (bf16_t)o1;
            ov[2] = (bf16_t)o2; ov[3] = (bf16_t)o3;
            const long m2 = rowBase + (long)i * 256;
            *(bf16x4*)(A2 + m2 * 4096 + colBase + d0) = ov;
        }
    }
}

// ---------------------------------------------------------------------------
extern "C" void kernel_launch(void* const* d_in, const int* in_sizes, int n_in,
                              void* d_out, int out_size, void* d_ws, size_t ws_size,
                              hipStream_t stream)
{
    const float* values  = (const float*)d_in[0];
    const float* keysI   = (const float*)d_in[1];
    const float* queries = (const float*)d_in[2];
    const int*   mask    = (const int*)  d_in[3];
    const float* Wv = (const float*)d_in[4];
    const float* bv = (const float*)d_in[5];
    const float* Wk = (const float*)d_in[6];
    const float* bk = (const float*)d_in[7];
    const float* Wq = (const float*)d_in[8];
    const float* bq = (const float*)d_in[9];
    const float* Wo = (const float*)d_in[10];
    const float* bo = (const float*)d_in[11];

    // Workspace: 4 x (8192 x 4096) bf16 = 256 MB
    const size_t projElems = (size_t)8192 * 4096;
    bf16_t* Qw = (bf16_t*)d_ws;
    bf16_t* Kw = Qw + projElems;
    bf16_t* Vw = Kw + projElems;
    bf16_t* A2 = Vw + projElems;

    dim3 blk(256, 1, 1);
    dim3 gproj(4096 / BN, 8192 / BM, 1);   // 32 x 64
    wmma_gemm<true, false><<<gproj, blk, 0, stream>>>(
        (const void*)queries, Wq, bq, (void*)Qw, 512, 4096);
    wmma_gemm<true, false><<<gproj, blk, 0, stream>>>(
        (const void*)keysI, Wk, bk, (void*)Kw, 512, 4096);
    wmma_gemm<true, false><<<gproj, blk, 0, stream>>>(
        (const void*)values, Wv, bv, (void*)Vw, 512, 4096);

    attn_kernel<<<8192 / 8, blk, 0, stream>>>(Qw, Kw, Vw, mask, A2);

    dim3 gout(512 / BN, 8192 / BM, 1);     // 4 x 64
    wmma_gemm<false, true><<<gout, blk, 0, stream>>>(
        (const void*)A2, Wo, bo, d_out, 4096, 512);
}
